// RKNSingleCell_64158221468311
// MI455X (gfx1250) — matrix operand order repaired and checked by reading, compiled, not run on hardware
//
#include <hip/hip_runtime.h>

#define B_TOT 8192
#define LOD   64
#define LSD   128
#define ORG   128
#define KC    16
#define IBLK  32

typedef __attribute__((ext_vector_type(16))) __bf16         v16bf;
typedef __attribute__((ext_vector_type(16))) unsigned short v16us;
typedef __attribute__((ext_vector_type(8)))  float          v8f;

typedef __attribute__((address_space(3))) float lds_f32;

// packed-weight layout inside workspace (ushort offsets)
#define WB_MEAN  0        // 64 x 128
#define WB_COVAR 8192     // 64 x 128
#define WB_NORM  16384    // 64 x 64
#define WB_COEF  20480    // 16 x 128
#define WB_TOTAL 22528

__device__ inline unsigned short f2bf(float x) {          // round-to-nearest-even
  unsigned int u = __float_as_uint(x);
  u += 0x7FFFu + ((u >> 16) & 1u);
  return (unsigned short)(u >> 16);
}

// pack 2 x f32 -> 2 x bf16 (truncate) in one v_perm_b32
__device__ inline unsigned int pk2bf(float lo, float hi) {
  return __builtin_amdgcn_perm(__float_as_uint(hi), __float_as_uint(lo), 0x07060302u);
}

__device__ inline v8f wmma_bf16(v16bf a, v16bf b, v8f c) {
  return __builtin_amdgcn_wmma_f32_16x16x32_bf16(false, a, false, b, (short)0, c, false, false);
}

// A-fragment (16x32 bf16) from an LDS row-major bf16 tile.
// lanes 0-15: row = lane, K 0..15
// lanes16-31: row = lane-16, K 16..31
__device__ inline v16bf load_A_lds(const unsigned short* base, int ld, int m0, int kbase, int lane) {
  const unsigned short* p = base + (m0 + (lane & 15)) * ld + kbase + ((lane >> 4) << 4);
  v16us u;
#pragma unroll
  for (int t = 0; t < 8; ++t) u[t] = p[t];
#pragma unroll
  for (int t = 0; t < 8; ++t) u[8 + t] = p[8 + t];
  return __builtin_bit_cast(v16bf, u);
}

// B-fragment (32x16 bf16), B[k,n] = W[n0+n, kbase+k], W pre-packed bf16 row-major NxK.
// lanes 0-15: N = lane, K 0..15 ; lanes 16-31: N = lane-16, K 16..31
// 16 contiguous ushorts -> two global_load_b128, zero VALU.
__device__ inline v16bf load_B_bf16(const unsigned short* Wb, int ldk, int n0, int kbase, int lane) {
  const unsigned short* p = Wb + (n0 + (lane & 15)) * ldk + kbase + ((lane >> 4) << 4);
  v16us u;
#pragma unroll
  for (int t = 0; t < 16; ++t) u[t] = p[t];
  return __builtin_bit_cast(v16bf, u);
}

// ---------------------------------------------------------------------------
// Kernel 0: one-shot weight pack f32 -> bf16 (RNE). 22528 elements, L2-resident.
// ---------------------------------------------------------------------------
__global__ void k_pack(const float* __restrict__ Wm, const float* __restrict__ Wc,
                       const float* __restrict__ Wn, const float* __restrict__ Wk,
                       unsigned short* __restrict__ wb) {
  int t = blockIdx.x * 256 + threadIdx.x;   // grid covers 8192
  if (t < 8192) {
    wb[WB_MEAN + t]  = f2bf(Wm[t]);
    wb[WB_COVAR + t] = f2bf(Wc[t]);
  }
  if (t < 4096) wb[WB_NORM + t] = f2bf(Wn[t]);
  if (t < 2048) wb[WB_COEF + t] = f2bf(Wk[t]);
}

// ---------------------------------------------------------------------------
// Kernel 1: all dense GEMMs on WMMA + fused epilogues.
// 128 blocks x 128 threads (4 waves). Each wave owns 16 batch rows.
// ---------------------------------------------------------------------------
__global__ __launch_bounds__(128) void k_gemm(
    const float* __restrict__ input,  const float* __restrict__ state,
    const unsigned short* __restrict__ wb,
    const float* __restrict__ b_mean, const float* __restrict__ b_covar,
    const float* __restrict__ b_norm, const float* __restrict__ b_coef,
    float* __restrict__ ws_obs, float* __restrict__ ws_c,
    float* __restrict__ ws_coeff, float* __restrict__ ws_partial)
{
  __shared__ unsigned short inA[64 * 136];   // input tile, bf16, padded stride
  __shared__ unsigned short pmA[64 * 136];   // post_mean tile, bf16
  __shared__ unsigned short hA [64 * 72];    // h tile, bf16
  __shared__ float logitsS[64 * 16];
  __shared__ float redS[4];

  const int tid  = threadIdx.x;
  const int lane = tid & 31;
  const int wave = tid >> 5;
  const int row0 = blockIdx.x * 64;

  __builtin_prefetch(wb, 0, 0);

  // stage + convert A tiles: 2 elements per iteration via v_perm packing
  for (int idx = tid * 2; idx < 64 * 128; idx += 256) {
    int r = idx >> 7, c = idx & 127;
    const float* ip = input + (size_t)(row0 + r) * ORG + c;
    const float* sp = state + (size_t)(row0 + r) * 320 + c;
    *(unsigned int*)&inA[r * 136 + c] = pk2bf(ip[0], ip[1]);
    *(unsigned int*)&pmA[r * 136 + c] = pk2bf(sp[0], sp[1]);
  }
  __syncthreads();

  const int m0  = wave * 16;
  const int nlo = lane & 15;
  const int mhi = (lane >> 4) * 8;   // C-frag: m = mhi + r, n = nlo + 16*nt
  const v8f vz = {};

  // ---- h = input @ W_mean^T  and  c = input @ W_covar^T (shared A-frags) ----
  v8f hacc[4], cacc[4];
#pragma unroll
  for (int nt = 0; nt < 4; ++nt) { hacc[nt] = vz; cacc[nt] = vz; }
#pragma unroll
  for (int kb = 0; kb < 4; ++kb) {
    v16bf a = load_A_lds(inA, 136, m0, kb * 32, lane);
#pragma unroll
    for (int nt = 0; nt < 4; ++nt) {
      hacc[nt] = wmma_bf16(a, load_B_bf16(wb + WB_MEAN,  ORG, nt * 16, kb * 32, lane), hacc[nt]);
      cacc[nt] = wmma_bf16(a, load_B_bf16(wb + WB_COVAR, ORG, nt * 16, kb * 32, lane), cacc[nt]);
    }
  }
#pragma unroll
  for (int nt = 0; nt < 4; ++nt) {
    int n = nt * 16 + nlo;
    float bn = b_mean[n];
#pragma unroll
    for (int r = 0; r < 8; ++r)
      hA[(m0 + mhi + r) * 72 + n] = f2bf(hacc[nt][r] + bn);
  }
  float asum = 0.f;
#pragma unroll
  for (int nt = 0; nt < 4; ++nt) {
    int n = nt * 16 + nlo;
    float bn = b_covar[n];
#pragma unroll
    for (int r = 0; r < 8; ++r) {
      float v = cacc[nt][r] + bn;
      ws_c[(size_t)(row0 + m0 + mhi + r) * LOD + n] = v;
      asum += fabsf(v);
    }
  }

  // ---- obs_mean = elu(h @ W_norm^T + b_norm) + 1 ----
  v8f oacc[4];
#pragma unroll
  for (int nt = 0; nt < 4; ++nt) oacc[nt] = vz;
#pragma unroll
  for (int kb = 0; kb < 2; ++kb) {
    v16bf a = load_A_lds(hA, 72, m0, kb * 32, lane);
#pragma unroll
    for (int nt = 0; nt < 4; ++nt)
      oacc[nt] = wmma_bf16(a, load_B_bf16(wb + WB_NORM, LOD, nt * 16, kb * 32, lane), oacc[nt]);
  }
#pragma unroll
  for (int nt = 0; nt < 4; ++nt) {
    int n = nt * 16 + nlo;
    float bn = b_norm[n];
#pragma unroll
    for (int r = 0; r < 8; ++r) {
      float v = oacc[nt][r] + bn;
      v = (v > 0.f) ? (v + 1.f) : __expf(v);           // elu(v)+1
      ws_obs[(size_t)(row0 + m0 + mhi + r) * LOD + n] = v;
    }
  }

  // ---- coeff = softmax(post_mean @ W_coef^T + b_coef) ----
  v8f lacc = vz;
#pragma unroll
  for (int kb = 0; kb < 4; ++kb)
    lacc = wmma_bf16(load_A_lds(pmA, 136, m0, kb * 32, lane),
                     load_B_bf16(wb + WB_COEF, LSD, 0, kb * 32, lane), lacc);
  {
    float bc = b_coef[nlo];
#pragma unroll
    for (int r = 0; r < 8; ++r)
      logitsS[(m0 + mhi + r) * 16 + nlo] = lacc[r] + bc;
  }

  // deterministic wave32 reduction of |c| partials
#pragma unroll
  for (int s = 16; s > 0; s >>= 1) asum += __shfl_xor(asum, s, 32);
  if (lane == 0) redS[wave] = asum;
  __syncthreads();
  if (tid == 0) ws_partial[blockIdx.x] = (redS[0] + redS[1]) + (redS[2] + redS[3]);

  if (lane < 16) {                 // one softmax row per lane (16 rows/wave)
    const float* L = &logitsS[(m0 + lane) * 16];
    float mx = L[0];
#pragma unroll
    for (int k = 1; k < 16; ++k) mx = fmaxf(mx, L[k]);
    float e[16], s = 0.f;
#pragma unroll
    for (int k = 0; k < 16; ++k) { e[k] = __expf(L[k] - mx); s += e[k]; }
    float inv = 1.f / s;
    float* dst = &ws_coeff[(size_t)(row0 + m0 + lane) * 16];
#pragma unroll
    for (int k = 0; k < 16; ++k) dst[k] = e[k] * inv;
  }
}

// ---------------------------------------------------------------------------
// Kernel 1b: deterministic grid-wide sum of |c| partials -> 1/S
// ---------------------------------------------------------------------------
__global__ void k_reduce(const float* __restrict__ partial, float* __restrict__ invS) {
  if (threadIdx.x == 0) {
    float s = 0.f;
    for (int i = 0; i < 128; ++i) s += partial[i];
    *invS = 1.0f / s;
  }
}

// ---------------------------------------------------------------------------
// Kernel 2: banded transition + Kalman update. One thread per batch row,
// each block covers 128 batch rows and half of the i-range.
// Basis gathered into LDS via GLOBAL_LOAD_ASYNC_TO_LDS (ASYNCcnt tracked).
// ---------------------------------------------------------------------------
__global__ __launch_bounds__(128) void k_band(
    const float* __restrict__ state,
    const float* __restrict__ tm11, const float* __restrict__ tm12,
    const float* __restrict__ tm21, const float* __restrict__ tm22,
    const float* __restrict__ ltc,
    const float* __restrict__ ws_obs, const float* __restrict__ ws_c,
    const float* __restrict__ ws_coeff, const float* __restrict__ invSp,
    float* __restrict__ out)
{
  __shared__ float bas[4 * IBLK * 6 * KC];   // [blk][ii][d][k], 48KB

  const int tid = threadIdx.x;
  const int bb  = blockIdx.x >> 1;           // batch group (128 rows)
  const int ilo = (blockIdx.x & 1) * IBLK;   // i-range half

  // cooperative gather of band-masked basis (j = i + d - 3) into LDS.
  // valid entries: async DMA global->LDS; out-of-band entries: DS-pipe zeros
  // (disjoint addresses, so no async/DS ordering hazard before the barrier).
  for (int idx = tid; idx < 4 * IBLK * 6 * KC; idx += 128) {
    int k    = idx & 15;
    int rest = idx >> 4;
    int d    = rest % 6;
    int q    = rest / 6;
    int ii   = q % IBLK;
    int blk  = q / IBLK;
    int i = ilo + ii;
    int j = i + d - 3;
    if (j >= 0 && j < 64) {
      const float* src = (blk == 0) ? tm11 : (blk == 1) ? tm12 : (blk == 2) ? tm21 : tm22;
      const float* gp = src + ((size_t)k * 64 + i) * 64 + j;
#if defined(__gfx1250__)
      unsigned int lds_off = (unsigned int)(unsigned long long)(lds_f32*)&bas[idx];
      unsigned long long ga = (unsigned long long)(size_t)gp;
      asm volatile("global_load_async_to_lds_b32 %0, %1, off"
                   :: "v"(lds_off), "v"(ga) : "memory");
#else
      bas[idx] = *gp;
#endif
    } else {
      bas[idx] = 0.f;
    }
  }
#if defined(__gfx1250__)
  asm volatile("s_wait_asynccnt 0x0" ::: "memory");
#endif
  __syncthreads();

  const int b = bb * 128 + tid;
  const float* sb = state + (size_t)b * 320;
  float* ob = out + (size_t)b * 320;
  const float invS = *invSp;

  float coeff[16];
#pragma unroll
  for (int k = 0; k < 16; ++k) coeff[k] = ws_coeff[(size_t)b * 16 + k];

  // sliding window over j = i-3 .. i+2 (zeros outside [0,64) kill those terms)
  float wpmu[6], wpml[6], wcu[6], wcl[6], wcs[6];
#pragma unroll
  for (int t = 0; t < 6; ++t) {
    int j = ilo - 3 + t;
    bool v = (j >= 0 && j < 64);               // uniform branch
    wpmu[t] = v ? sb[j]       : 0.f;
    wpml[t] = v ? sb[64 + j]  : 0.f;
    wcu[t]  = v ? sb[128 + j] : 0.f;
    wcl[t]  = v ? sb[192 + j] : 0.f;
    wcs[t]  = v ? sb[256 + j] : 0.f;
  }

  for (int ii = 0; ii < IBLK; ++ii) {
    int i = ilo + ii;
    float pu = 0.f, pl = 0.f, a_pcu = 0.f, a_pcl = 0.f, a_pcs = 0.f;
#pragma unroll
    for (int d = 0; d < 6; ++d) {
      const float* p11 = &bas[((0 * IBLK + ii) * 6 + d) * KC];
      const float* p12 = &bas[((1 * IBLK + ii) * 6 + d) * KC];
      const float* p21 = &bas[((2 * IBLK + ii) * 6 + d) * KC];
      const float* p22 = &bas[((3 * IBLK + ii) * 6 + d) * KC];
      float e11 = 0.f, e12 = 0.f, e21 = 0.f, e22 = 0.f;
#pragma unroll
      for (int k = 0; k < 16; ++k) {    // LDS broadcast reads, b128-vectorizable
        float ck = coeff[k];
        e11 += ck * p11[k]; e12 += ck * p12[k];
        e21 += ck * p21[k]; e22 += ck * p22[k];
      }
      float pmu = wpmu[d], pml = wpml[d];
      float cu = wcu[d], cl = wcl[d], cs = wcs[d];
      pu += e11 * pmu + e12 * pml;
      pl += e21 * pmu + e22 * pml;
      a_pcu += e11 * e11 * cu + 2.f * e11 * e12 * cs + e12 * e12 * cl;
      a_pcl += e21 * e21 * cu + 2.f * e21 * e22 * cs + e22 * e22 * cl;
      a_pcs += e21 * e11 * cu + (e22 * e11 + e21 * e12) * cs + e22 * e12 * cl;
    }
    // advance window: drop j=i-3, append j=i+3
#pragma unroll
    for (int t = 0; t < 5; ++t) {
      wpmu[t] = wpmu[t + 1]; wpml[t] = wpml[t + 1];
      wcu[t] = wcu[t + 1];   wcl[t] = wcl[t + 1];   wcs[t] = wcs[t + 1];
    }
    {
      int j = i + 3;
      bool v = (j < 64);                         // uniform branch
      wpmu[5] = v ? sb[j]       : 0.f;
      wpml[5] = v ? sb[64 + j]  : 0.f;
      wcu[5]  = v ? sb[128 + j] : 0.f;
      wcl[5]  = v ? sb[192 + j] : 0.f;
      wcs[5]  = v ? sb[256 + j] : 0.f;
    }
    float li = ltc[i], lj = ltc[64 + i];
    float tcu = (li > 0.f) ? (li + 1.f) : __expf(li);   // elu+1
    float tcl = (lj > 0.f) ? (lj + 1.f) : __expf(lj);
    float pcu = a_pcu + tcu;
    float pcl = a_pcl + tcl;
    float pcs = a_pcs;
    float ocov  = ws_c[(size_t)b * LOD + i] * invS;
    float denom = pcu + ocov;
    float qu = pcu / denom, ql = pcs / denom;
    float resid = ws_obs[(size_t)b * LOD + i] - pu;
    float cf = 1.f - qu;
    ob[i]       = pu + qu * resid;
    ob[64 + i]  = pl + ql * resid;
    ob[128 + i] = cf * pcu;
    ob[192 + i] = pcl - ql * pcs;
    ob[256 + i] = cf * pcs;
  }
}

// ---------------------------------------------------------------------------
extern "C" void kernel_launch(void* const* d_in, const int* in_sizes, int n_in,
                              void* d_out, int out_size, void* d_ws, size_t ws_size,
                              hipStream_t stream) {
  (void)in_sizes; (void)n_in; (void)out_size; (void)ws_size;
  const float* input   = (const float*)d_in[0];
  const float* state   = (const float*)d_in[1];
  const float* W_mean  = (const float*)d_in[2];
  const float* b_mean  = (const float*)d_in[3];
  const float* W_covar = (const float*)d_in[4];
  const float* b_covar = (const float*)d_in[5];
  const float* W_norm  = (const float*)d_in[6];
  const float* b_norm  = (const float*)d_in[7];
  const float* W_coef  = (const float*)d_in[8];
  const float* b_coef  = (const float*)d_in[9];
  const float* tm11    = (const float*)d_in[10];
  const float* tm12    = (const float*)d_in[11];
  const float* tm21    = (const float*)d_in[12];
  const float* tm22    = (const float*)d_in[13];
  const float* ltc     = (const float*)d_in[14];
  float* out = (float*)d_out;

  float* ws        = (float*)d_ws;
  float* ws_obs    = ws;                       // 8192*64
  float* ws_c      = ws + 524288;              // 8192*64
  float* ws_coeff  = ws + 1048576;             // 8192*16
  float* ws_part   = ws + 1179648;             // 128
  float* ws_invS   = ws + 1179776;             // 1
  unsigned short* ws_wb = (unsigned short*)(ws + 1179780);  // 22528 bf16

  k_pack<<<32, 256, 0, stream>>>(W_mean, W_covar, W_norm, W_coef, ws_wb);
  k_gemm<<<B_TOT / 64, 128, 0, stream>>>(input, state, ws_wb,
                                         b_mean, b_covar, b_norm, b_coef,
                                         ws_obs, ws_c, ws_coeff, ws_part);
  k_reduce<<<1, 32, 0, stream>>>(ws_part, ws_invS);
  k_band<<<(B_TOT / 128) * 2, 128, 0, stream>>>(state, tm11, tm12, tm21, tm22, ltc,
                                                ws_obs, ws_c, ws_coeff, ws_invS, out);
}
